// EdgeNetwork_77257871720572
// MI455X (gfx1250) — matrix operand order repaired and checked by reading, not compiled, not run on hardware
//
#include <hip/hip_runtime.h>
#include <hip/hip_bf16.h>
#include <stdint.h>

typedef __attribute__((ext_vector_type(16))) __bf16   v16bf;
typedef __attribute__((ext_vector_type(8)))  float    v8f;
typedef __attribute__((ext_vector_type(8)))  uint32_t v8u;
typedef int v4i __attribute__((vector_size(16)));   // matches async-builtin pointee

#define D        64
#define BFEAT    16
#define KTILES   34      // (16*64 + 64) / 32 : 16 bond blocks + 1 bias block
#define CTILES   4       // 64 output cols / 16
#define NSTRIDE  68      // padded LDS row stride in floats (bank-conflict avoidance)
#define BSTRIDE  20      // bond row stride: 16 features + slot16 = 1.0f (bias), 16B-aligned

#define AS1 __attribute__((address_space(1)))
#define AS3 __attribute__((address_space(3)))

#if defined(__has_builtin)
#  if __has_builtin(__builtin_amdgcn_global_load_async_to_lds_b128)
#    define HAVE_ASYNC_LDS 1
#  endif
#endif
#ifndef HAVE_ASYNC_LDS
#  define HAVE_ASYNC_LDS 0
#endif

#if defined(__has_builtin) && __has_builtin(__builtin_amdgcn_s_wait_asynccnt)
#  define WAIT_ASYNC() __builtin_amdgcn_s_wait_asynccnt(0)
#else
#  define WAIT_ASYNC() asm volatile("s_wait_asynccnt 0" ::: "memory")
#endif

// ---------------- zero the output (harness poisons d_out) ----------------
__global__ void zero_out_kernel(float4* __restrict__ out, int n4) {
  int i = blockIdx.x * blockDim.x + threadIdx.x;
  if (i < n4) out[i] = make_float4(0.f, 0.f, 0.f, 0.f);
}

// ---------------- pack kernel+bias into bf16 B-fragment layout ----------------
// bw[(((kt*4 + ct)*32 + lane)*8 + v)] holds the bf16 pair (h=0 lo, h=1 hi).
// B matrix (32x16, KxN) per-lane mapping: N = lane%16, K = 2*(v&3) + h + 8*(lane>=16) + 16*(v>>2).
__device__ __forceinline__ uint32_t pack2_bf16(float lo, float hi) {
  uint32_t ulo = __float_as_uint(lo);
  uint32_t uhi = __float_as_uint(hi);
  ulo = (ulo + 0x7FFFu + ((ulo >> 16) & 1u)) >> 16;   // round-to-nearest-even
  uhi = (uhi + 0x7FFFu + ((uhi >> 16) & 1u)) >> 16;
  return (uhi << 16) | (ulo & 0xFFFFu);
}

__global__ void pack_B_kernel(const float* __restrict__ kern,
                              const float* __restrict__ bias,
                              uint32_t* __restrict__ bw) {
  int fid = blockIdx.x * blockDim.x + threadIdx.x;
  const int total = KTILES * CTILES * 32 * 8;
  if (fid >= total) return;
  const int v    = fid & 7;
  const int lane = (fid >> 3) & 31;
  const int ct   = (fid >> 8) & 3;
  const int kt   = fid >> 10;
  const int half = lane >> 4;
  const int lm   = lane & 15;
  const int i    = ct * 16 + lm;                 // output column
  float f[2];
#pragma unroll
  for (int h = 0; h < 2; ++h) {
    const int kl = 2 * (v & 3) + h + 8 * half + 16 * (v >> 2);        // K within 32-tile (B layout)
    const int k  = kt * 32 + kl;                 // global K in [0,1088)
    const int b  = k >> 6;                       // bond feature block
    const int jj = k & 63;                       // j within 64
    f[h] = (b < BFEAT) ? kern[b * (D * D) + i * D + jj] : bias[i * D + jj];
  }
  bw[fid] = pack2_bf16(f[0], f[1]);
}

// ---------------- fused edge-network kernel ----------------
// Per WG: 64 edges. transformed[64x64] = A(64x1088) @ K'(1088x64), A built on the fly
// as bond[e,b]*n[e,j]; bias block uses coefficient 1.0 stored in bond slot 16.
__global__ __launch_bounds__(256)
void edge_network_kernel(const float* __restrict__ atom,
                         const float* __restrict__ bond,
                         const int*  __restrict__ pairs,
                         const uint32_t* __restrict__ bw,
                         float* __restrict__ out,
                         int nEdges) {
  __shared__ float s_n[64 * NSTRIDE];
  __shared__ float s_bond[64 * BSTRIDE];
  __shared__ int   s_src[64];
  __shared__ int   s_dst[64];

  const int t  = threadIdx.x;
  const int e0 = blockIdx.x * 64;
  const bool fullBlock = (e0 + 64 <= nEdges);

  if (t < 64) {
    const int e = e0 + t;
    int2 pi = make_int2(0, 0);
    if (e < nEdges) pi = ((const int2*)pairs)[e];
    s_src[t] = pi.x;
    s_dst[t] = pi.y;
  }
  __syncthreads();

  const int elq = t >> 2;   // edge this thread stages (4 threads per edge)
  const int q   = t & 3;    // quarter of the 64-float row / 16-float bond row

#if HAVE_ASYNC_LDS
  if (fullBlock) {
    // Async gather: per-lane global source, per-lane LDS destination (ASYNCcnt).
    const float* arow = atom + (size_t)s_dst[elq] * D + q * 16;
    float*       nrow = &s_n[elq * NSTRIDE + q * 16];
#pragma unroll
    for (int i = 0; i < 4; ++i) {
      __builtin_amdgcn_global_load_async_to_lds_b128(
          (AS1 v4i*)(uintptr_t)(arow + i * 4),
          (AS3 v4i*)(void*)(nrow + i * 4), 0, 0);
    }
    __builtin_amdgcn_global_load_async_to_lds_b128(
        (AS1 v4i*)(uintptr_t)(bond + (size_t)(e0 + elq) * BFEAT + q * 4),
        (AS3 v4i*)(void*)&s_bond[elq * BSTRIDE + q * 4], 0, 0);
    WAIT_ASYNC();
  } else
#endif
  {
    const int  e     = e0 + elq;
    const bool valid = (e < nEdges);
    const float4 z = make_float4(0.f, 0.f, 0.f, 0.f);
    const float4* arow = (const float4*)(atom + (size_t)s_dst[elq] * D) + q * 4;
    float4* nrow = (float4*)(&s_n[elq * NSTRIDE + q * 16]);
#pragma unroll
    for (int i = 0; i < 4; ++i) {
      float4 val = z;
      if (valid) val = arow[i];
      nrow[i] = val;
    }
    float4 bv = z;
    if (valid) bv = ((const float4*)(bond + (size_t)e * BFEAT))[q];
    *(float4*)(&s_bond[elq * BSTRIDE + q * 4]) = bv;
  }
  if (t < 64) s_bond[t * BSTRIDE + BFEAT] = 1.0f;   // bias-block coefficient
  __syncthreads();

  const int lane = t & 31;
  const int w    = t >> 5;
  const int eg   = w & 3;            // edge group (16 edges)
  const int ctb  = (w >> 2) * 2;     // this wave handles col tiles ctb, ctb+1
  const int half = lane >> 4;
  const int lm   = lane & 15;
  const int el   = eg * 16 + lm;     // edge row (A-matrix M = lane%16)

  v8f acc0 = {};
  v8f acc1 = {};
  const float* nrow = &s_n[el * NSTRIDE];
  const float* brow = &s_bond[el * BSTRIDE];

  // Per-lane base of this wave's B-fragment stream.
  // Strides (dwords): +256 per col tile, +1024 per k-step.
  const uint32_t* bwp = bw + ((size_t)ctb * 32 + lane) * 8;

  __builtin_prefetch(bwp, 0, 3);     // one-shot warm-up (global_prefetch_b8)

  // One WMMA step: build A fragment (bond[b]*n slice, bf16) and do 2 matmuls.
  auto step = [&](int kt, v8u pb0, v8u pb1) {
    const int   bidx = kt >> 1;              // 0..16; slot 16 == 1.0f (bias block)
    const int   j0   = (kt & 1) * 32;
    const float s    = brow[bidx];
    // A 16x32 bf16 layout: lane takes j = j0+8*half+[0,8) (VGPR0-3) and
    // j0+16+8*half+[0,8) (VGPR4-7). n-row loads are loop-invariant -> in VGPRs.
    const float4* r0 = (const float4*)(nrow + j0 + 8 * half);
    const float4* r1 = (const float4*)(nrow + j0 + 16 + 8 * half);
    const float4 a0 = r0[0], a1 = r0[1];
    const float4 b0 = r1[0], b1 = r1[1];
    const float run[16] = {a0.x, a0.y, a0.z, a0.w, a1.x, a1.y, a1.z, a1.w,
                           b0.x, b0.y, b0.z, b0.w, b1.x, b1.y, b1.z, b1.w};
    v16bf afrag;
#pragma unroll
    for (int i = 0; i < 16; ++i) afrag[i] = (__bf16)(s * run[i]);
    acc0 = __builtin_amdgcn_wmma_f32_16x16x32_bf16(
        false, afrag, false, __builtin_bit_cast(v16bf, pb0), (short)0, acc0, false, false);
    acc1 = __builtin_amdgcn_wmma_f32_16x16x32_bf16(
        false, afrag, false, __builtin_bit_cast(v16bf, pb1), (short)0, acc1, false, false);
  };

  // Software pipeline: B fragments fetched one k-step ahead (separate regs),
  // so each WMMA has a full iteration of VALU between load issue and consume.
  v8u nb0 = *(const v8u*)(bwp);
  v8u nb1 = *(const v8u*)(bwp + 256);
#pragma unroll 2
  for (int kt = 0; kt < KTILES - 2; ++kt) {    // 32 steady iterations, no guards
    const v8u cb0 = nb0, cb1 = nb1;
    nb0 = *(const v8u*)(bwp + (size_t)(kt + 1) * 1024);
    nb1 = *(const v8u*)(bwp + (size_t)(kt + 1) * 1024 + 256);
    step(kt, cb0, cb1);
  }
  {                                            // epilogue: kt = 32, 33
    const v8u cb0 = nb0, cb1 = nb1;
    nb0 = *(const v8u*)(bwp + (size_t)33 * 1024);
    nb1 = *(const v8u*)(bwp + (size_t)33 * 1024 + 256);
    step(32, cb0, cb1);
    step(33, nb0, nb1);
  }

  // C/D layout: VGPR r -> M = r + 8*half, N = lane%16. Scatter-add to out[src].
#pragma unroll
  for (int r = 0; r < 8; ++r) {
    const int elr = eg * 16 + r + 8 * half;
    const int e   = e0 + elr;
    if (e < nEdges) {
      const size_t base = (size_t)s_src[elr] * D;
      unsafeAtomicAdd(&out[base + (ctb + 0) * 16 + lm], acc0[r]);
      unsafeAtomicAdd(&out[base + (ctb + 1) * 16 + lm], acc1[r]);
    }
  }
}

extern "C" void kernel_launch(void* const* d_in, const int* in_sizes, int n_in,
                              void* d_out, int out_size, void* d_ws, size_t ws_size,
                              hipStream_t stream) {
  const float* atom  = (const float*)d_in[0];   // (N, 64)
  const float* bond  = (const float*)d_in[1];   // (E, 16)
  const int*   pairs = (const int*)d_in[2];     // (E, 2)
  const float* kern  = (const float*)d_in[3];   // (16, 4096)
  const float* bias  = (const float*)d_in[4];   // (4096,)
  float* out = (float*)d_out;                   // (N, 64)

  const int nEdges = in_sizes[1] / BFEAT;

  const int n4 = out_size / 4;
  zero_out_kernel<<<(n4 + 255) / 256, 256, 0, stream>>>((float4*)out, n4);

  uint32_t* bw = (uint32_t*)d_ws;               // 139,264 bytes of packed bf16 B
  const int totalB = KTILES * CTILES * 32 * 8;
  pack_B_kernel<<<(totalB + 255) / 256, 256, 0, stream>>>(kern, bias, bw);

  const int nBlocks = (nEdges + 63) / 64;
  edge_network_kernel<<<nBlocks, 256, 0, stream>>>(atom, bond, pairs, bw, out, nEdges);
}
